// HeteroMPNNPredictor1TestNodeType_57303453663602
// MI455X (gfx1250) — matrix operand order repaired and checked by reading, not compiled, not run on hardware
//
#include <hip/hip_runtime.h>

// ---------------------------------------------------------------------------
// Hetero GCN (5-layer) for MI455X / gfx1250.
//  - GEMMs  (h @ W, K=N=128) via v_wmma_f32_16x16x32_f16 (f16 in, f32 acc)
//    with weights staged in LDS (padded rows -> conflict-free ds_load_b128)
//  - seg_mean via inv-degree-scaled global_atomic_add_f32 scatter
//  - bias + ReLU + residual fused in finalize
// ---------------------------------------------------------------------------

typedef __attribute__((ext_vector_type(16))) _Float16 v16h;
typedef __attribute__((ext_vector_type(8)))  float    v8f;
typedef __attribute__((ext_vector_type(4)))  float    v4f;

#define H 128
#define BROW 136          // padded LDS row stride in halves (272B = 68 dwords -> bank+4/row)
#define N_TEST_CONST 10000

// ----------------------------- utility kernels -----------------------------

__global__ void zero_f4(float4* __restrict__ p, int n4) {
    int i = blockIdx.x * blockDim.x + threadIdx.x;
    if (i < n4) p[i] = make_float4(0.f, 0.f, 0.f, 0.f);
}

__global__ void count_deg(const int* __restrict__ dst, float* __restrict__ deg, int n_edges) {
    int e = blockIdx.x * blockDim.x + threadIdx.x;
    if (e < n_edges) atomicAdd(&deg[dst[e]], 1.0f);
}

__global__ void invert_deg(float* __restrict__ deg, int n) {
    int i = blockIdx.x * blockDim.x + threadIdx.x;
    if (i < n) deg[i] = 1.0f / fmaxf(deg[i], 1.0f);
}

// W (KxN f32, row-major) -> Wt (NxK f16, row-major)  [128x128]
__global__ void prep_weight(const float* __restrict__ W, _Float16* __restrict__ Wt) {
    int idx = blockIdx.x * blockDim.x + threadIdx.x;  // 0..16383
    int n = idx >> 7, k = idx & 127;
    Wt[n * H + k] = (_Float16)W[k * H + n];
}

// ----------------------------- node encoding -------------------------------

// h_ast[n] = [ emb[label[n]] (64)  ||  content[n](256) @ Wc(256x64) + bc ]
__global__ void encode_ast(const int* __restrict__ label, const float* __restrict__ content,
                           const float* __restrict__ emb, const float* __restrict__ Wc,
                           const float* __restrict__ bc, float* __restrict__ hA, int n_ast) {
    int nid = blockIdx.x * 4 + (threadIdx.x >> 6);
    int j   = threadIdx.x & 63;
    if (nid >= n_ast) return;
    hA[nid * H + j] = emb[label[nid] * 64 + j];
    const float* c = content + (long long)nid * 256;
    float s = bc[j];
    #pragma unroll 4
    for (int k = 0; k < 256; ++k) s += c[k] * Wc[k * 64 + j];
    hA[nid * H + 64 + j] = s;
}

__global__ void init_test(const float* __restrict__ temb, float* __restrict__ hT, int total) {
    int i = blockIdx.x * blockDim.x + threadIdx.x;
    if (i < total) hT[i] = temb[i & (H - 1)];
}

// ----------------------------- WMMA GEMM  ----------------------------------
// C(Mx128 f32) = A(Mx128 f32, cvt->f16) * Bt^T   where Bt is (128x128 f16, Bt[n][k]=W[k][n]).
// 256 threads = 8 waves; block computes a 128x128 slab; Bt staged in padded LDS.
__global__ __launch_bounds__(256) void gemm_h128(const float* __restrict__ A,
                                                 const _Float16* __restrict__ Bt,
                                                 float* __restrict__ C, int M) {
    __shared__ alignas(32) _Float16 Bs[H * BROW];   // ~34 KB

    // ---- stage Bt into LDS: 2048 chunks of 16B, padded rows ----
    {
        typedef __attribute__((ext_vector_type(8))) _Float16 v8h;
        #pragma unroll
        for (int it = 0; it < 8; ++it) {
            int idx = it * 256 + threadIdx.x;       // 0..2047
            int r = idx >> 4, c = idx & 15;         // row, 16B-chunk within row
            *(v8h*)(Bs + r * BROW + c * 8) = *(const v8h*)(Bt + r * H + c * 8);
        }
    }
    __syncthreads();

    const int lane = threadIdx.x & 31;
    const int wave = threadIdx.x >> 5;
    const int m = lane & 15;   // tile row (A) / tile col (B,C)
    const int g = lane >> 4;   // lane-group -> K-half selector
    const int row0 = blockIdx.x * 128 + wave * 16;
    int row = row0 + m;
    int rowc = row < M ? row : (M - 1);   // clamp loads; stores are guarded
    const float* arow = A + (long long)rowc * H;

    v8f acc[8] = {};

    #pragma unroll
    for (int ks = 0; ks < 4; ++ks) {
        const int K0 = ks * 32;
        // A fragment (contiguous k order, same as B): lanes g=0 hold K0+0..15,
        // lanes g=1 hold K0+16..31; convert f32 -> f16 inline.
        v4f a0 = *(const v4f*)(arow + K0 + 16 * g);
        v4f a1 = *(const v4f*)(arow + K0 + 16 * g + 4);
        v4f a2 = *(const v4f*)(arow + K0 + 16 * g + 8);
        v4f a3 = *(const v4f*)(arow + K0 + 16 * g + 12);
        v16h afrag;
        #pragma unroll
        for (int j = 0; j < 4; ++j) {
            afrag[j]      = (_Float16)a0[j];
            afrag[4 + j]  = (_Float16)a1[j];
            afrag[8 + j]  = (_Float16)a2[j];
            afrag[12 + j] = (_Float16)a3[j];
        }
        #pragma unroll
        for (int nt = 0; nt < 8; ++nt) {
            // B fragment (32x16): lanes 0-15 hold K0+0..15, lanes 16-31 hold
            // K0+16..31 of column n = m; contiguous 32B from padded LDS row.
            v16h bfrag = *(const v16h*)(Bs + (nt * 16 + m) * BROW + K0 + 16 * g);
            acc[nt] = __builtin_amdgcn_wmma_f32_16x16x32_f16(
                false, afrag, false, bfrag, (short)0, acc[nt], false, false);
        }
    }

    // C layout: VGPR r -> M = r + 8*g, N = lane%16
    #pragma unroll
    for (int nt = 0; nt < 8; ++nt) {
        #pragma unroll
        for (int r = 0; r < 8; ++r) {
            int orow = row0 + 8 * g + r;
            if (orow < M) C[(long long)orow * H + nt * 16 + m] = acc[nt][r];
        }
    }
}

// ----------------------------- edge scatter --------------------------------
// One wave per edge: acc[dst] += t[src] * inv_deg[dst]   (128 floats, float4/lane)
__global__ void edge_scatter(const float* __restrict__ t, const int* __restrict__ src,
                             const int* __restrict__ dst, const float* __restrict__ inv_deg,
                             float* __restrict__ acc, int n_edges) {
    int e = (blockIdx.x * blockDim.x + threadIdx.x) >> 5;
    int lane = threadIdx.x & 31;
    if (e >= n_edges) return;
    int s = src[e], d = dst[e];
    float w = inv_deg[d];
    const float4 v = *(const float4*)(t + (long long)s * H + lane * 4);
    float* a = acc + (long long)d * H + lane * 4;
    atomicAdd(a + 0, v.x * w);
    atomicAdd(a + 1, v.y * w);
    atomicAdd(a + 2, v.z * w);
    atomicAdd(a + 3, v.w * w);
}

// out = relu(acc + bias[col]) (+ res)   elementwise over n*128
__global__ void finalize_h(const float* __restrict__ acc, const float* __restrict__ bias,
                           const float* __restrict__ res, float* __restrict__ out, int total) {
    int i = blockIdx.x * blockDim.x + threadIdx.x;
    if (i >= total) return;
    float v = fmaxf(acc[i] + bias[i & (H - 1)], 0.0f);
    if (res) v += res[i];
    out[i] = v;
}

// ----------------------------- decode + softmax ----------------------------
// One wave per node: logits = h @ decW(128x2) + b; out = [logits | softmax]
__global__ void decode(const float* __restrict__ h, const float* __restrict__ W,
                       const float* __restrict__ b, float* __restrict__ out, int n) {
    int node = (blockIdx.x * blockDim.x + threadIdx.x) >> 5;
    int lane = threadIdx.x & 31;
    if (node >= n) return;
    const float4 v = *(const float4*)(h + (long long)node * H + lane * 4);
    int k0 = lane * 4;
    float s0 = v.x * W[(k0 + 0) * 2] + v.y * W[(k0 + 1) * 2] +
               v.z * W[(k0 + 2) * 2] + v.w * W[(k0 + 3) * 2];
    float s1 = v.x * W[(k0 + 0) * 2 + 1] + v.y * W[(k0 + 1) * 2 + 1] +
               v.z * W[(k0 + 2) * 2 + 1] + v.w * W[(k0 + 3) * 2 + 1];
    #pragma unroll
    for (int off = 16; off >= 1; off >>= 1) {
        s0 += __shfl_xor(s0, off, 32);
        s1 += __shfl_xor(s1, off, 32);
    }
    if (lane == 0) {
        s0 += b[0]; s1 += b[1];
        float mx = fmaxf(s0, s1);
        float e0 = expf(s0 - mx), e1 = expf(s1 - mx);
        float inv = 1.0f / (e0 + e1);
        out[(long long)node * 2 + 0] = s0;
        out[(long long)node * 2 + 1] = s1;
        out[(long long)n * 2 + (long long)node * 2 + 0] = e0 * inv;
        out[(long long)n * 2 + (long long)node * 2 + 1] = e1 * inv;
    }
}

// ----------------------------- host driver ---------------------------------

static inline unsigned cdiv_u(long long a, long long b) { return (unsigned)((a + b - 1) / b); }

extern "C" void kernel_launch(void* const* d_in, const int* in_sizes, int n_in,
                              void* d_out, int out_size, void* d_ws, size_t ws_size,
                              hipStream_t stream) {
    const int*   ast_label   = (const int*)  d_in[0];
    const float* ast_content = (const float*)d_in[1];
    const int*   src_aa = (const int*)d_in[2];
    const int*   dst_aa = (const int*)d_in[3];
    const int*   src_at = (const int*)d_in[4];
    const int*   dst_at = (const int*)d_in[5];
    const int*   src_ta = (const int*)d_in[6];
    const int*   dst_ta = (const int*)d_in[7];
    /* d_in[8] = n_test scalar on device; fixed by problem definition */
    const float* emb      = (const float*)d_in[9];
    const float* Wc       = (const float*)d_in[10];
    const float* bc       = (const float*)d_in[11];
    const float* test_emb = (const float*)d_in[12];
    const float* W_aa     = (const float*)d_in[13];
    const float* W_at     = (const float*)d_in[14];
    const float* W_ta     = (const float*)d_in[15];
    const float* b_ast    = (const float*)d_in[16];
    const float* b_test   = (const float*)d_in[17];
    const float* dec_W    = (const float*)d_in[18];
    const float* dec_b    = (const float*)d_in[19];

    const int N_AST  = in_sizes[0];
    const int N_TEST = N_TEST_CONST;
    const int E_AA = in_sizes[2], E_AT = in_sizes[4], E_TA = in_sizes[6];

    // ---- workspace bump allocator ----
    char* ws = (char*)d_ws;
    size_t off = 0;
    auto walloc = [&](size_t bytes) -> void* {
        off = (off + 255) & ~(size_t)255;
        void* p = ws + off;
        off += bytes;
        return p;
    };
    const size_t szA = (size_t)N_AST  * H * sizeof(float);
    const size_t szT = (size_t)N_TEST * H * sizeof(float);

    float* hA0      = (float*)walloc(szA);
    float* hA1      = (float*)walloc(szA);
    float* hT0      = (float*)walloc(szT);
    float* hT1      = (float*)walloc(szT);
    float* t_aa     = (float*)walloc(szA);
    float* t_at     = (float*)walloc(szA);
    float* t_ta     = (float*)walloc(szT);
    float* acc_ast  = (float*)walloc(szA);
    float* acc_test = (float*)walloc(szT);
    float* inv_aa   = (float*)walloc((size_t)N_AST  * sizeof(float));
    float* inv_ta   = (float*)walloc((size_t)N_AST  * sizeof(float));
    float* inv_at   = (float*)walloc((size_t)N_TEST * sizeof(float));
    _Float16* Wt_aa = (_Float16*)walloc((size_t)H * H * sizeof(_Float16));
    _Float16* Wt_at = (_Float16*)walloc((size_t)H * H * sizeof(_Float16));
    _Float16* Wt_ta = (_Float16*)walloc((size_t)H * H * sizeof(_Float16));
    (void)ws_size; (void)n_in;

    const int TB = 256;

    // ---- degrees (constant across layers) ----
    zero_f4<<<cdiv_u(N_AST / 4, TB), TB, 0, stream>>>((float4*)inv_aa, N_AST / 4);
    zero_f4<<<cdiv_u(N_AST / 4, TB), TB, 0, stream>>>((float4*)inv_ta, N_AST / 4);
    zero_f4<<<cdiv_u(N_TEST / 4, TB), TB, 0, stream>>>((float4*)inv_at, N_TEST / 4);
    count_deg<<<cdiv_u(E_AA, TB), TB, 0, stream>>>(dst_aa, inv_aa, E_AA);
    count_deg<<<cdiv_u(E_TA, TB), TB, 0, stream>>>(dst_ta, inv_ta, E_TA);
    count_deg<<<cdiv_u(E_AT, TB), TB, 0, stream>>>(dst_at, inv_at, E_AT);
    invert_deg<<<cdiv_u(N_AST, TB), TB, 0, stream>>>(inv_aa, N_AST);
    invert_deg<<<cdiv_u(N_AST, TB), TB, 0, stream>>>(inv_ta, N_AST);
    invert_deg<<<cdiv_u(N_TEST, TB), TB, 0, stream>>>(inv_at, N_TEST);

    // ---- node encoding ----
    encode_ast<<<cdiv_u(N_AST, 4), 256, 0, stream>>>(ast_label, ast_content, emb, Wc, bc, hA0, N_AST);
    init_test<<<cdiv_u((long long)N_TEST * H, TB), TB, 0, stream>>>(test_emb, hT0, N_TEST * H);

    // ---- layers ----
    float* hA_cur = hA0; float* hA_nxt = hA1;
    float* hT_cur = hT0; float* hT_nxt = hT1;
    const float* resA = nullptr; const float* resT = nullptr;

    for (int L = 0; L < 5; ++L) {
        const size_t woff = (size_t)L * H * H;
        prep_weight<<<64, TB, 0, stream>>>(W_aa + woff, Wt_aa);
        prep_weight<<<64, TB, 0, stream>>>(W_at + woff, Wt_at);
        prep_weight<<<64, TB, 0, stream>>>(W_ta + woff, Wt_ta);

        zero_f4<<<cdiv_u((long long)N_AST * H / 4, TB), TB, 0, stream>>>((float4*)acc_ast, N_AST * H / 4);
        zero_f4<<<cdiv_u((long long)N_TEST * H / 4, TB), TB, 0, stream>>>((float4*)acc_test, N_TEST * H / 4);

        gemm_h128<<<cdiv_u(N_AST, 128), 256, 0, stream>>>(hA_cur, Wt_aa, t_aa, N_AST);
        gemm_h128<<<cdiv_u(N_AST, 128), 256, 0, stream>>>(hA_cur, Wt_at, t_at, N_AST);
        gemm_h128<<<cdiv_u(N_TEST, 128), 256, 0, stream>>>(hT_cur, Wt_ta, t_ta, N_TEST);

        edge_scatter<<<cdiv_u((long long)E_AA * 32, TB), TB, 0, stream>>>(t_aa, src_aa, dst_aa, inv_aa, acc_ast, E_AA);
        edge_scatter<<<cdiv_u((long long)E_TA * 32, TB), TB, 0, stream>>>(t_ta, src_ta, dst_ta, inv_ta, acc_ast, E_TA);
        edge_scatter<<<cdiv_u((long long)E_AT * 32, TB), TB, 0, stream>>>(t_at, src_at, dst_at, inv_at, acc_test, E_AT);

        const float* ra = (L == 1 || L == 3) ? resA : nullptr;
        const float* rt = (L == 1 || L == 3) ? resT : nullptr;
        finalize_h<<<cdiv_u((long long)N_AST * H, TB), TB, 0, stream>>>(acc_ast, b_ast + L * H, ra, hA_nxt, N_AST * H);
        finalize_h<<<cdiv_u((long long)N_TEST * H, TB), TB, 0, stream>>>(acc_test, b_test + L * H, rt, hT_nxt, N_TEST * H);

        if (L == 0 || L == 2) { resA = hA_nxt; resT = hT_nxt; }
        { float* tmp = hA_cur; hA_cur = hA_nxt; hA_nxt = tmp; }
        { float* tmp = hT_cur; hT_cur = hT_nxt; hT_nxt = tmp; }
    }

    // ---- decode ----
    decode<<<cdiv_u((long long)N_AST * 32, TB), TB, 0, stream>>>(hA_cur, dec_W, dec_b, (float*)d_out, N_AST);
    (void)out_size;
}
